// NCCLoss_19275813224853
// MI455X (gfx1250) — matrix operand order repaired and checked by reading, not compiled, not run on hardware
//
#include <hip/hip_runtime.h>

typedef __attribute__((ext_vector_type(16))) _Float16 v16h;
typedef __attribute__((ext_vector_type(2)))  _Float16 h2;
typedef __attribute__((ext_vector_type(8)))  float    v8f;

#define VDIM    192
#define BRICK   16
#define HALO    4
#define REG     24                      // BRICK + 2*HALO
#define NB      (VDIM / BRICK)          // 12
#define NBLK    (NB * NB * NB)          // 1728
#define EPS_NCC 1e-5f

#define RAW_ELEMS   (REG * REG * REG)       // 13824
#define T1_FSTRIDE  (REG * BRICK * REG)     // 9216  : [z:24][x:16][y:24]
#define T2_FSTRIDE  (BRICK * BRICK * REG)   // 6144  : [y:16][x:16][z:24]

// LDS layout (bytes). t2 aliases the raw region (dead after stage 1).
#define LDS_SI_OFF   0                       // 13824 f16
#define LDS_SJ_OFF   27648                   // 13824 f16
#define LDS_T1_OFF   61440                   // 5*9216 f16  -> ends 153600
#define LDS_T2_OFF   0                       // 5*6144 f16  -> ends 61440
#define LDS_ACC_OFF  153600                  // 8 floats (per-wave partials)
#define LDS_BYTES    153664

// Constant banded box-filter weights in WMMA B-operand layout (32x16 f16):
// lane holds column n = lane&15; element i holds K = (i<8 ? i : i+8) + (lane>=16 ? 8 : 0).
// B[k][n] = 1/9 for n <= k <= n+8 (output n consumes inputs n..n+8), else 0.
__device__ __forceinline__ v16h make_bfrag(int lane) {
  const int n  = lane & 15;
  const int kb = (lane & 16) ? 8 : 0;
  v16h b;
#pragma unroll
  for (int i = 0; i < 16; ++i) {
    int k = i + ((i >= 8) ? 8 : 0) + kb;
    float w = (k >= n && k <= n + 8) ? (1.0f / 9.0f) : 0.0f;
    b[i] = (_Float16)w;
  }
  return b;
}

// A-operand (16-bit 16x32): element pair (2v, 2v+1) of the v16h holds
// K = {k0, k0+1}, k0 = (v<4 ? 2v : 2v+8) + (lane>=16 ? 8 : 0).  k0 is always
// even, so with the filtered axis contiguous in LDS each pair is one aligned
// 32-bit LDS load.
__device__ __forceinline__ int a_k0(int v, int lane) {
  return ((v < 4) ? (2 * v) : (2 * v + 8)) + ((lane & 16) ? 8 : 0);
}

__global__ __launch_bounds__(256)
void ncc_brick_kernel(const float* __restrict__ I,   // target
                      const float* __restrict__ J,   // predict
                      float* __restrict__ partial)
{
  extern __shared__ char lds[];
  _Float16* sI   = (_Float16*)(lds + LDS_SI_OFF);
  _Float16* sJ   = (_Float16*)(lds + LDS_SJ_OFF);
  _Float16* t1   = (_Float16*)(lds + LDS_T1_OFF);
  _Float16* t2   = (_Float16*)(lds + LDS_T2_OFF);
  float*    sWav = (float*)(lds + LDS_ACC_OFF);

  const h2* sI2 = (const h2*)sI;
  const h2* sJ2 = (const h2*)sJ;
  const h2* t1p = (const h2*)t1;
  const h2* t2p = (const h2*)t2;

  const int tid  = threadIdx.x;
  const int lane = tid & 31;
  const int wid  = tid >> 5;

  // ---- load 24^3 halo region of I and J (zero-padded), f32 -> f16 ----
  const int b  = blockIdx.x;
  const int bz = b / (NB * NB);
  const int by = (b / NB) % NB;
  const int bx = b % NB;
  const int z0 = bz * BRICK - HALO;
  const int y0 = by * BRICK - HALO;
  const int x0 = bx * BRICK - HALO;

  for (int i = tid; i < RAW_ELEMS; i += 256) {
    int lz = i / (REG * REG);
    int ly = (i / REG) % REG;
    int lx = i % REG;
    int gz = z0 + lz, gy = y0 + ly, gx = x0 + lx;
    float vi = 0.0f, vj = 0.0f;
    if ((unsigned)gz < VDIM && (unsigned)gy < VDIM && (unsigned)gx < VDIM) {
      int g = (gz * VDIM + gy) * VDIM + gx;
      vi = I[g];
      vj = J[g];
    }
    sI[i] = (_Float16)vi;
    sJ[i] = (_Float16)vj;
  }
  __syncthreads();

  const v16h bfrag = make_bfrag(lane);
  const int  row   = lane & 15;           // A-matrix row held by this lane
  const int  nCol  = lane & 15;           // C/D column held by this lane
  const int  mofs  = (lane & 16) ? 8 : 0; // C/D row offset for this lane half

  // ---- stage 1: 9-tap box along x (5 fields x 36 line-tiles) ----
  // lines = (z,y) in 24x24; A(line, k=x) = field(z,y,x); D cols = x_out.
  for (int j = wid; j < 5 * 36; j += 8) {
    int f = j / 36, t = j % 36;
    v16h a;
    int base2 = (t * 16 + row) * (REG / 2);   // h2 index of raw row base
#pragma unroll
    for (int v = 0; v < 8; ++v) {
      int k0 = a_k0(v, lane);
      h2 val = {(_Float16)0.0f, (_Float16)0.0f};
      if (k0 < REG) {
        h2 pi = sI2[base2 + (k0 >> 1)];
        h2 pj = sJ2[base2 + (k0 >> 1)];
        val = (f == 0) ? pi
            : (f == 1) ? pj
            : (f == 2) ? pi * pi
            : (f == 3) ? pj * pj
                       : pi * pj;
      }
      a[2 * v]     = val[0];
      a[2 * v + 1] = val[1];
    }
    v8f c = {};
    c = __builtin_amdgcn_wmma_f32_16x16x32_f16(false, a, false, bfrag,
                                               (short)0, c, false, false);
    // D row r+mofs -> line t*16+r+mofs = (z,y); store t1[z][x_out][y].
    int line0 = t * 16 + mofs;
    int zc = line0 / REG, yc = line0 % REG;
#pragma unroll
    for (int r = 0; r < 8; ++r) {
      t1[f * T1_FSTRIDE + (zc * 16 + nCol) * REG + yc] = (_Float16)c[r];
      ++yc;
      if (yc == REG) { yc = 0; ++zc; }
    }
  }
  __syncthreads();

  // ---- stage 2: 9-tap box along y (5 fields x 24 z-slices) ----
  // lines = x (16) within slice z=t; A(line, k=y) = t1[z][x][y]; D cols = y_out.
  for (int j = wid; j < 5 * 24; j += 8) {
    int f = j / 24, zt = j % 24;
    v16h a;
    int base2 = (f * T1_FSTRIDE + (zt * 16 + row) * REG) >> 1;
#pragma unroll
    for (int v = 0; v < 8; ++v) {
      int k0 = a_k0(v, lane);
      h2 val = {(_Float16)0.0f, (_Float16)0.0f};
      if (k0 < REG) val = t1p[base2 + (k0 >> 1)];
      a[2 * v]     = val[0];
      a[2 * v + 1] = val[1];
    }
    v8f c = {};
    c = __builtin_amdgcn_wmma_f32_16x16x32_f16(false, a, false, bfrag,
                                               (short)0, c, false, false);
#pragma unroll
    for (int r = 0; r < 8; ++r) {
      int x = r + mofs;
      t2[f * T2_FSTRIDE + (nCol * 16 + x) * REG + zt] = (_Float16)c[r];
    }
  }
  __syncthreads();

  // ---- stage 3: 9-tap box along z + cc, fused (16 y-slices) ----
  // lines = x (16) within slice y=t; A(line, k=z) = t2[y][x][z]; D cols = z_out.
  float csum = 0.0f;
  for (int t = wid; t < 16; t += 8) {
    v8f acc[5];
#pragma unroll
    for (int f = 0; f < 5; ++f) {
      v16h a;
      int base2 = (f * T2_FSTRIDE + (t * 16 + row) * REG) >> 1;
#pragma unroll
      for (int v = 0; v < 8; ++v) {
        int k0 = a_k0(v, lane);
        h2 val = {(_Float16)0.0f, (_Float16)0.0f};
        if (k0 < REG) val = t2p[base2 + (k0 >> 1)];
        a[2 * v]     = val[0];
        a[2 * v + 1] = val[1];
      }
      v8f c = {};
      acc[f] = __builtin_amdgcn_wmma_f32_16x16x32_f16(false, a, false, bfrag,
                                                      (short)0, c, false, false);
    }
#pragma unroll
    for (int r = 0; r < 8; ++r) {
      float mu1 = acc[0][r];
      float mu2 = acc[1][r];
      float s1  = acc[2][r] - mu1 * mu1;
      float s2  = acc[3][r] - mu2 * mu2;
      float s12 = acc[4][r] - mu1 * mu2;
      csum += (s12 * s12) / (s1 * s2 + EPS_NCC);
    }
  }

  // ---- deterministic block reduction ----
#pragma unroll
  for (int off = 16; off > 0; off >>= 1)
    csum += __shfl_down(csum, off, 32);
  if (lane == 0) sWav[wid] = csum;
  __syncthreads();
  if (tid == 0) {
    float s = 0.0f;
#pragma unroll
    for (int w = 0; w < 8; ++w) s += sWav[w];
    partial[b] = s;
  }
}

__global__ __launch_bounds__(256)
void ncc_finalize_kernel(const float* __restrict__ partial,
                         float* __restrict__ out)
{
  __shared__ float red[256];
  float s = 0.0f;
  for (int i = threadIdx.x; i < NBLK; i += 256) s += partial[i];
  red[threadIdx.x] = s;
  __syncthreads();
  for (int w = 128; w > 0; w >>= 1) {
    if ((int)threadIdx.x < w) red[threadIdx.x] += red[threadIdx.x + w];
    __syncthreads();
  }
  if (threadIdx.x == 0) {
    const float invN = 1.0f / ((float)VDIM * (float)VDIM * (float)VDIM);
    out[0] = 1.0f - red[0] * invN;
  }
}

extern "C" void kernel_launch(void* const* d_in, const int* in_sizes, int n_in,
                              void* d_out, int out_size, void* d_ws, size_t ws_size,
                              hipStream_t stream) {
  (void)in_sizes; (void)n_in; (void)out_size; (void)ws_size;
  const float* predict = (const float*)d_in[0];
  const float* target  = (const float*)d_in[1];
  float* out     = (float*)d_out;
  float* partial = (float*)d_ws;   // NBLK floats of scratch

  // 150 KB dynamic LDS per block (gfx1250 WGP has 320 KB) — opt in.
  hipFuncSetAttribute((const void*)ncc_brick_kernel,
                      hipFuncAttributeMaxDynamicSharedMemorySize, LDS_BYTES);

  ncc_brick_kernel<<<dim3(NBLK), dim3(256), LDS_BYTES, stream>>>(
      target, predict, partial);
  ncc_finalize_kernel<<<dim3(1), dim3(256), 0, stream>>>(partial, out);
}
